// RationaleSelectorModel_59227599012006
// MI455X (gfx1250) — compile-verified, matrix operand order checked by reading, not run on hardware
//
#include <hip/hip_runtime.h>
#include <hip/hip_bf16.h>
#include <math.h>

// Problem constants (match reference)
#define BATCH 32
#define TLEN  4096
#define DDIM  768
#define HDIM  1024
#define KT_N  24      // DDIM / 32  (K tiles of 32 for bf16 WMMA)
#define NT_N  64      // HDIM / 16  (N tiles of 16)
#define NB    8       // N tiles per chunk
#define NCHUNK 8      // NT_N / NB

typedef __attribute__((ext_vector_type(16))) __bf16 v16bf;
typedef __attribute__((ext_vector_type(8)))  float  v8f;

// K offset inside a 32-wide K tile for element i (0..15) of a v16bf fragment,
// per the CDNA5 16-bit A/B layout:
//   VGPR0..3 hold K={0..7}, VGPR4..7 hold K={16..23} for lanes 0..15;
//   lanes 16..31 hold the same rows with K offset by +8.
__device__ __host__ inline int frag_koff(int i) {
  int v = i >> 1, odd = i & 1;
  return (v < 4) ? (v * 2 + odd) : (16 + (v - 4) * 2 + odd);
}

// ---------------------------------------------------------------------------
// Kernel 0: W1 [768 x 1024] f32 row-major -> bf16 B-fragments in d_ws.
// Layout: w1f[((nt*KT_N + kt)*32 + lane)*16 + i]
// so each lane of the GEMM loads one contiguous 32B v16bf per (nt,kt).
// ---------------------------------------------------------------------------
__global__ __launch_bounds__(256)
void convert_w1(const float* __restrict__ W1, unsigned short* __restrict__ w1f_raw) {
  __bf16* w1f = (__bf16*)w1f_raw;
  int f = blockIdx.x * 256 + threadIdx.x;          // < 64*24*32*16 = 786432
  int i    = f & 15;
  int lane = (f >> 4) & 31;
  int kt   = (f >> 9) % KT_N;
  int nt   = f / (16 * 32 * KT_N);
  int col  = lane & 15;                            // N within tile
  int j    = frag_koff(i) + ((lane >> 4) ? 8 : 0); // K within 32-tile
  int k    = kt * 32 + j;
  w1f[f] = (__bf16)W1[k * HDIM + nt * 16 + col];
}

// ---------------------------------------------------------------------------
// Kernel 1: fused  mask -> LayerNorm -> (bf16 WMMA GEMM with W1) -> +b1 ->
//           erf-GELU -> dot(W2) + b2 -> attn mask  => scores [B*T]
// One wave per 16-row tile. 8 waves / block, 24KB LDS per wave (bf16 A frags).
// ---------------------------------------------------------------------------
__global__ __launch_bounds__(256)
void fused_score(const float* __restrict__ emb,  const float* __restrict__ attn,
                 const float* __restrict__ lng,  const float* __restrict__ lnb,
                 const unsigned short* __restrict__ w1f_raw,
                 const float* __restrict__ b1,   const float* __restrict__ W2,
                 const float* __restrict__ b2,   float* __restrict__ scores) {
  extern __shared__ __bf16 lds[];
  const __bf16* w1f = (const __bf16*)w1f_raw;
  const int lane = threadIdx.x & 31;
  const int wave = threadIdx.x >> 5;
  __bf16* afrag = lds + wave * (16 * DDIM);        // 12288 bf16 = 24KB per wave

  const int  tile    = blockIdx.x * 8 + wave;      // 1024*8 = 8192 tiles exactly
  const long rowbase = (long)tile * 16;

  // ---- Pass 1: per-row LayerNorm stats (coalesced loads + shfl reductions)
  float mu_r = 0.f, rs_r = 0.f, am_r = 0.f;        // stats for row (lane&15)
  for (int r = 0; r < 16; ++r) {
    long rg = rowbase + r;
    int  bb = (int)(rg >> 12);                     // T = 4096
    int  tt = (int)(rg & (TLEN - 1));
    float av = attn[bb * TLEN + tt];
    const float* row = emb + rg * DDIM;
    float s1 = 0.f, s2 = 0.f;
    #pragma unroll
    for (int it = 0; it < 24; ++it) {
      float x = row[lane + 32 * it] * av;
      s1 += x; s2 += x * x;
    }
    #pragma unroll
    for (int m = 16; m >= 1; m >>= 1) {
      s1 += __shfl_xor(s1, m, 32);
      s2 += __shfl_xor(s2, m, 32);
    }
    float mu   = s1 * (1.f / DDIM);
    float var  = s2 * (1.f / DDIM) - mu * mu;
    float rstd = rsqrtf(var + 1e-5f);
    if ((lane & 15) == r) { mu_r = mu; rs_r = rstd; am_r = av; }
  }

  // ---- Pass 2: normalize + write bf16 A fragments in WMMA lane order
  {
    long rg = rowbase + (lane & 15);
    const float* row = emb + rg * DDIM;
    int jbase = (lane >> 4) * 8;
    for (int kt = 0; kt < KT_N; ++kt) {
      __bf16* dst = afrag + kt * 512 + lane * 16;
      #pragma unroll
      for (int i = 0; i < 16; ++i) {
        int k = kt * 32 + jbase + frag_koff(i);
        float x  = row[k] * am_r;
        float xn = (x - mu_r) * rs_r * lng[k] + lnb[k];
        dst[i] = (__bf16)xn;
      }
    }
  }
  __syncthreads();

  // ---- Pass 3: GEMM (wmma bf16) + fused GELU + W2 reduction
  float sacc[8];
  #pragma unroll
  for (int v = 0; v < 8; ++v) sacc[v] = 0.f;
  const v8f vzero = {0.f, 0.f, 0.f, 0.f, 0.f, 0.f, 0.f, 0.f};

  for (int ch = 0; ch < NCHUNK; ++ch) {
    v8f acc[NB];
    #pragma unroll
    for (int nt = 0; nt < NB; ++nt) acc[nt] = vzero;

    for (int kt = 0; kt < KT_N; ++kt) {
      v16bf a = *(const v16bf*)(afrag + kt * 512 + lane * 16);
      const __bf16* bbase = w1f + ((long)(ch * NB) * KT_N + kt) * 512 + lane * 16;
      #pragma unroll
      for (int nt = 0; nt < NB; ++nt) {
        v16bf bm = *(const v16bf*)(bbase + (long)nt * KT_N * 512);
        acc[nt] = __builtin_amdgcn_wmma_f32_16x16x32_bf16(
            false, a, false, bm, (short)0, acc[nt], false, false);
      }
    }
    // epilogue: h1 = acc + b1 ; gelu(erf) ; score partial += gelu * W2
    #pragma unroll
    for (int nt = 0; nt < NB; ++nt) {
      int col = (ch * NB + nt) * 16 + (lane & 15);
      float b1v = b1[col];
      float w2v = W2[col];
      #pragma unroll
      for (int v = 0; v < 8; ++v) {
        float h  = acc[nt][v] + b1v;
        float ge = 0.5f * h * (1.0f + erff(h * 0.70710678118f));
        sacc[v] += ge * w2v;
      }
    }
  }

  // reduce over the 16 N-lanes of each half-wave
  #pragma unroll
  for (int m = 8; m >= 1; m >>= 1)
    #pragma unroll
    for (int v = 0; v < 8; ++v) sacc[v] += __shfl_xor(sacc[v], m, 32);

  if ((lane & 15) == 0) {                          // lanes 0 (rows 0-7) and 16 (rows 8-15)
    float b2v = b2[0];
    int hf = lane >> 4;
    for (int v = 0; v < 8; ++v) {
      long rg = rowbase + hf * 8 + v;
      int  bb = (int)(rg >> 12), tt = (int)(rg & (TLEN - 1));
      float av = attn[bb * TLEN + tt];
      float s  = sacc[v] + b2v;
      scores[rg] = (av == 0.f) ? -1e9f : s;
    }
  }
}

// ---------------------------------------------------------------------------
// Kernel 2: per-batch-row entmax-1.5 + rank masks. One 1024-thread block per
// batch row; bitonic sort (desc) + Hillis-Steele cumsums in LDS (~56KB).
// ---------------------------------------------------------------------------
__global__ __launch_bounds__(1024)
void entmax_topk(const float* __restrict__ scores, const float* __restrict__ attn,
                 float* __restrict__ out) {
  __shared__ float key[TLEN];
  __shared__ unsigned short kidx[TLEN];
  __shared__ float cx[TLEN];
  __shared__ float cx2[TLEN];
  __shared__ float s_teff;
  __shared__ int   s_supp;

  const int b = blockIdx.x, tid = threadIdx.x;
  if (tid == 0) { s_teff = 0.f; s_supp = 1; }
  __syncthreads();

  float tpart = 0.f;
  for (int p = tid; p < TLEN; p += 1024) {
    key[p]  = scores[b * TLEN + p] * 0.5f;     // /TAU(=1), then x/2 for entmax1.5
    kidx[p] = (unsigned short)p;
    tpart  += attn[b * TLEN + p];
  }
  atomicAdd(&s_teff, tpart);
  __syncthreads();

  // bitonic sort, descending
  for (int size = 2; size <= TLEN; size <<= 1) {
    for (int stride = size >> 1; stride > 0; stride >>= 1) {
      for (int i = tid; i < TLEN / 2; i += 1024) {
        int lo = 2 * i - (i & (stride - 1));
        int hi = lo + stride;
        bool up = (lo & size) != 0;            // final full pass -> descending
        float a = key[lo], c = key[hi];
        if ((a < c) != up) {
          key[lo] = c; key[hi] = a;
          unsigned short t = kidx[lo]; kidx[lo] = kidx[hi]; kidx[hi] = t;
        }
      }
      __syncthreads();
    }
  }

  // cumsum(x), cumsum(x^2) via Hillis-Steele (4 elems / thread)
  for (int p = tid; p < TLEN; p += 1024) { float k0 = key[p]; cx[p] = k0; cx2[p] = k0 * k0; }
  __syncthreads();
  for (int off = 1; off < TLEN; off <<= 1) {
    float a0[4], a1[4];
    #pragma unroll
    for (int e = 0; e < 4; ++e) {
      int p = tid + e * 1024;
      a0[e] = (p >= off) ? cx[p - off]  : 0.f;
      a1[e] = (p >= off) ? cx2[p - off] : 0.f;
    }
    __syncthreads();
    #pragma unroll
    for (int e = 0; e < 4; ++e) {
      int p = tid + e * 1024;
      cx[p] += a0[e]; cx2[p] += a1[e];
    }
    __syncthreads();
  }

  // tau per prefix, support size
  for (int p = tid; p < TLEN; p += 1024) {
    float rho  = (float)(p + 1);
    float mean = cx[p] / rho;
    float msq  = cx2[p] / rho;
    float ss   = rho * (msq - mean * mean);
    float dlt  = (1.f - ss) / rho;
    float tau  = mean - sqrtf(fmaxf(dlt, 0.f));
    if (tau <= key[p]) atomicMax(&s_supp, p + 1);
    cx[p] = tau;                               // own-slot overwrite only
  }
  __syncthreads();

  const float tau_star = cx[s_supp - 1];
  const float teff = s_teff;
  const float rhos[5] = {0.1f, 0.3f, 0.5f, 0.7f, 0.9f};
  int kk[5];
  #pragma unroll
  for (int r = 0; r < 5; ++r) {
    int k0 = (int)floorf(rhos[r] * teff);
    kk[r] = k0 < 1 ? 1 : k0;
  }

  for (int p = tid; p < TLEN; p += 1024) {
    int   o  = kidx[p];                        // original position; p = inv_rank
    float av = attn[b * TLEN + o];
    float z  = key[p] - tau_star;
    z = (z > 0.f) ? z * z : 0.f;
    z *= av;
    out[b * TLEN + o] = z;                     // channel 0: z
    #pragma unroll
    for (int r = 0; r < 5; ++r)                // channels 1..5: g == h (value)
      out[(r + 1) * (BATCH * TLEN) + b * TLEN + o] = (p < kk[r]) ? av : 0.f;
  }
}

// ---------------------------------------------------------------------------
extern "C" void kernel_launch(void* const* d_in, const int* in_sizes, int n_in,
                              void* d_out, int out_size, void* d_ws, size_t ws_size,
                              hipStream_t stream) {
  const float* emb  = (const float*)d_in[0];
  const float* attn = (const float*)d_in[1];
  const float* ln_g = (const float*)d_in[2];
  const float* ln_b = (const float*)d_in[3];
  const float* W1   = (const float*)d_in[4];
  const float* b1   = (const float*)d_in[5];
  const float* W2   = (const float*)d_in[6];
  const float* b2   = (const float*)d_in[7];
  // d_in[8] = ids (unused by the math)

  unsigned short* w1f = (unsigned short*)d_ws;                       // 1.5 MB bf16
  float* scores = (float*)((char*)d_ws + (size_t)DDIM * HDIM * 2);   // 512 KB f32

  convert_w1<<<(DDIM * HDIM) / 256, 256, 0, stream>>>(W1, w1f);
  // 8192 M-tiles, 8 waves/block, 24KB LDS per wave -> 192KB dynamic LDS
  fused_score<<<1024, 256, 8 * 16 * DDIM * sizeof(__bf16), stream>>>(
      emb, attn, ln_g, ln_b, w1f, b1, W2, b2, scores);
  entmax_topk<<<BATCH, 1024, 0, stream>>>(scores, attn, (float*)d_out);
}
